// SpectralConv1d_53824530154101
// MI455X (gfx1250) — matrix operand order repaired and checked by reading, not compiled, not run on hardware
//
#include <hip/hip_runtime.h>
#include <hip/hip_bf16.h>

typedef __attribute__((ext_vector_type(16))) _Float16 v16h;
typedef __attribute__((ext_vector_type(8)))  _Float16 v8h;
typedef __attribute__((ext_vector_type(8)))  float    v8f;
typedef __attribute__((ext_vector_type(4)))  float    v4f;

#define NLEN 8192              // L
#define NMASK 8191
#define NMODES 128             // MODES1
#define NROWS 8192             // B*Cin == B*Cout

// ---- workspace layout (bytes) ----
#define NORMX_OFF  (0)                          // 8192 f32
#define RINVO_OFF  (32*1024)                    // 8192 f32
#define BA_OFF     (64*1024)                    // 256*128*32 f16 = 2 MB
#define BC_OFF     (BA_OFF + 2*1024*1024)       // 4*8192*32 f16 = 2 MB
#define X1H_OFF    (BC_OFF + 2*1024*1024)       // 128*8192 f16  = 2 MB
#define WP_OFF     (X1H_OFF + 2*1024*1024)      // 128*4*128*32 f16 = 4 MB
#define WM_OFF     (WP_OFF + 4*1024*1024)       // 4 MB
#define MIXH_OFF   (WM_OFF + 4*1024*1024)       // 8192*128 f16 = 2 MB

// M[k][n] = (1 - sin(w k)) cos(w (k n mod N)) - cos(w k) sin(w (k n mod N))
__device__ __forceinline__ float mval(int k, int n) {
    const float w = 6.28318530717958647692f / 8192.0f;
    float th = w * (float)k;
    int kn = (k * n) & NMASK;              // k*n < 2^20, fits int
    float ph = w * (float)kn;
    return (1.0f - __sinf(th)) * __cosf(ph) - __cosf(th) * __sinf(ph);
}

__device__ __forceinline__ v16h cat8(v8h lo, v8h hi) {
    return __builtin_shufflevector(lo, hi, 0,1,2,3,4,5,6,7,8,9,10,11,12,13,14,15);
}

__device__ __forceinline__ v16h cvt16(v4f f0, v4f f1, v4f f2, v4f f3) {
    v16h a;
    a[0]=(_Float16)f0[0];  a[1]=(_Float16)f0[1];  a[2]=(_Float16)f0[2];  a[3]=(_Float16)f0[3];
    a[4]=(_Float16)f1[0];  a[5]=(_Float16)f1[1];  a[6]=(_Float16)f1[2];  a[7]=(_Float16)f1[3];
    a[8]=(_Float16)f2[0];  a[9]=(_Float16)f2[1];  a[10]=(_Float16)f2[2]; a[11]=(_Float16)f2[3];
    a[12]=(_Float16)f3[0]; a[13]=(_Float16)f3[1]; a[14]=(_Float16)f3[2]; a[15]=(_Float16)f3[3];
    return a;
}

// ---------- stage 0a: BA[nblk(256)][mode(128)][ni(32)] = M[mode][nblk*32+ni] (f16) ----------
__global__ __launch_bounds__(256) void k_build_BA(_Float16* __restrict__ BA) {
    int gid = blockIdx.x * 256 + threadIdx.x;          // 1,048,576 total
    int ni   = gid & 31;
    int mode = (gid >> 5) & 127;
    int nblk = gid >> 12;
    BA[gid] = (_Float16)mval(mode, nblk * 32 + ni);
}

// ---------- stage 0b: BC[kblk(4)][n(8192)][ki(32)] = M[n][kblk*32+ki] (f16) ----------
__global__ __launch_bounds__(256) void k_build_BC(_Float16* __restrict__ BC) {
    int gid = blockIdx.x * 256 + threadIdx.x;          // 1,048,576 total
    int ki   = gid & 31;
    int n    = (gid >> 5) & NMASK;
    int kblk = gid >> 18;
    BC[gid] = (_Float16)mval(n, kblk * 32 + ki);
}

// ---------- stage 0c: Wp/Wm[k][iblk(4)][o(128)][ii(32)] ----------
__global__ __launch_bounds__(256) void k_build_W(const float* __restrict__ w1,
                                                 _Float16* __restrict__ Wp,
                                                 _Float16* __restrict__ Wm) {
    int gid = blockIdx.x * 256 + threadIdx.x;          // 2,097,152 total
    int k = gid >> 14;
    int i = (gid >> 7) & 127;
    int o = gid & 127;
    int kb = (NMODES - k) & 127;
    float a  = w1[((size_t)i * 128 + o) * 128 + k];
    float bb = w1[((size_t)i * 128 + o) * 128 + kb];
    size_t idx = (((size_t)k * 4 + (i >> 5)) * 128 + o) * 32 + (i & 31);
    Wp[idx] = (_Float16)(a + bb);
    Wm[idx] = (_Float16)(a - bb);
}

// ---------- stage 1: per-input-row norm via Gram closed form ----------
// ||Mx||^2 = N*S0 + (N/2)*(S1 - S2), S1 = sum x[n]x[(N-n)%N], S2 = sum x[n]x[(N-2-n)%N]
__global__ __launch_bounds__(256) void k_norms(const float* __restrict__ x,
                                               float* __restrict__ rinvx) {
    __shared__ float srow[NLEN];
    __shared__ float acc[3];
    int row = blockIdx.x;
    int t = threadIdx.x;
    const float* base = x + (size_t)row * NLEN;
    for (int i = t; i < NLEN; i += 256) srow[i] = base[i];
    if (t < 3) acc[t] = 0.0f;
    __syncthreads();
    float s0 = 0.f, s1 = 0.f, s2 = 0.f;
    for (int n = t; n < NLEN; n += 256) {
        float v = srow[n];
        s0 += v * v;
        s1 += v * srow[(NLEN - n) & NMASK];
        s2 += v * srow[(NLEN - 2 - n) & NMASK];
    }
    atomicAdd(&acc[0], s0);
    atomicAdd(&acc[1], s1);
    atomicAdd(&acc[2], s2);
    __syncthreads();
    if (t == 0) {
        float nrm2 = 8192.0f * acc[0] + 4096.0f * (acc[1] - acc[2]);
        nrm2 = fmaxf(nrm2, 0.0f);
        float nrm = sqrtf(nrm2);
        rinvx[row] = (nrm > 0.0f) ? (1.0f / nrm) : 0.0f;
    }
}

// ---------- stage 2: X1h[k][row] = (M x)[k] / ||Mx||  for k<128, f16 ----------
// 16-row tile per WG (grid 512), one 16-mode N-tile per wave, software-pipelined K loop.
__global__ __launch_bounds__(256) void k_stageA(const float* __restrict__ x,
                                                const _Float16* __restrict__ BA,
                                                const float* __restrict__ rinvx,
                                                _Float16* __restrict__ X1h) {
    int wave = threadIdx.x >> 5;        // N-tile 0..7 (mode strip)
    int lane = threadIdx.x & 31;
    int half = lane >> 4;
    int l    = lane & 15;
    int rowbase = blockIdx.x * 16;
    const float* xrow = x + (size_t)(rowbase + l) * NLEN + half * 8;
    const _Float16* bp = BA + (size_t)(wave * 16 + l) * 32 + half * 8;

    // prologue loads (kk = 0)
    v4f f0 = *(const v4f*)(xrow + 0);
    v4f f1 = *(const v4f*)(xrow + 4);
    v4f f2 = *(const v4f*)(xrow + 16);
    v4f f3 = *(const v4f*)(xrow + 20);
    v16h bcur = cat8(*(const v8h*)bp, *(const v8h*)(bp + 16));

    v8f acc = {};
    for (int kk = 0; kk < NLEN / 32; ++kk) {
        int kn = (kk + 1 < NLEN / 32) ? (kk + 1) : (NLEN / 32 - 1);
        const float* an = xrow + kn * 32;
        __builtin_prefetch(an + 64, 0, 0);
        // issue next iteration's loads before consuming current fragments
        v4f nf0 = *(const v4f*)(an + 0);
        v4f nf1 = *(const v4f*)(an + 4);
        v4f nf2 = *(const v4f*)(an + 16);
        v4f nf3 = *(const v4f*)(an + 20);
        v16h bnext = cat8(*(const v8h*)(bp + (size_t)kn * 4096), *(const v8h*)(bp + (size_t)kn * 4096 + 16));   // 128*32 halves per K-block
        v16h a = cvt16(f0, f1, f2, f3);
        acc = __builtin_amdgcn_wmma_f32_16x16x32_f16(
            false, a, false, bcur, (short)0, acc, false, false);
        f0 = nf0; f1 = nf1; f2 = nf2; f3 = nf3;
        bcur = bnext;
    }
#pragma unroll
    for (int e = 0; e < 8; ++e) {
        int row  = rowbase + e + half * 8;
        int mode = wave * 16 + l;
        X1h[(size_t)mode * NROWS + row] = (_Float16)(acc[e] * rinvx[row]);
    }
}

// ---------- stage 3: mixing, one WG per mode k ----------
// mixed[b,o,k] = 0.5*( X1[k] @ Wp[k] + X1[k~] @ Wm[k] ),  M=64(b), N=128(o), K=128(i)
__global__ __launch_bounds__(256) void k_stageB(const _Float16* __restrict__ X1h,
                                                const _Float16* __restrict__ Wp,
                                                const _Float16* __restrict__ Wm,
                                                _Float16* __restrict__ mixed_h) {
    int k  = blockIdx.x;
    int kb = (NMODES - k) & 127;
    int wave = threadIdx.x >> 5;
    int lane = threadIdx.x & 31;
    int half = lane >> 4;
    int l    = lane & 15;
    int mt = wave >> 1;
    int nh = wave & 1;

    const _Float16* A1 = X1h + (size_t)k  * NROWS + (mt * 16 + l) * 128;
    const _Float16* A2 = X1h + (size_t)kb * NROWS + (mt * 16 + l) * 128;

    v8f acc[4] = {};
#pragma unroll
    for (int kk = 0; kk < 4; ++kk) {
        int c0 = kk * 32 + half * 8;
        v16h a1 = cat8(*(const v8h*)(A1 + c0), *(const v8h*)(A1 + c0 + 16));
        v16h a2 = cat8(*(const v8h*)(A2 + c0), *(const v8h*)(A2 + c0 + 16));
#pragma unroll
        for (int j = 0; j < 4; ++j) {
            int o = (nh * 4 + j) * 16 + l;
            v16h b1 = cat8(*(const v8h*)(Wp + (((size_t)k * 4 + kk) * 128 + o) * 32 + half * 8), *(const v8h*)(Wp + (((size_t)k * 4 + kk) * 128 + o) * 32 + half * 8 + 16));
            v16h b2 = cat8(*(const v8h*)(Wm + (((size_t)k * 4 + kk) * 128 + o) * 32 + half * 8), *(const v8h*)(Wm + (((size_t)k * 4 + kk) * 128 + o) * 32 + half * 8 + 16));
            acc[j] = __builtin_amdgcn_wmma_f32_16x16x32_f16(
                false, a1, false, b1, (short)0, acc[j], false, false);
            acc[j] = __builtin_amdgcn_wmma_f32_16x16x32_f16(
                false, a2, false, b2, (short)0, acc[j], false, false);
        }
    }
#pragma unroll
    for (int j = 0; j < 4; ++j) {
        int o = (nh * 4 + j) * 16 + l;
#pragma unroll
        for (int e = 0; e < 8; ++e) {
            int bg = mt * 16 + e + half * 8;         // batch index 0..63
            size_t row = (size_t)bg * 128 + o;       // output row = b*Cout + o
            mixed_h[row * NMODES + k] = (_Float16)(0.5f * acc[j][e]);
        }
    }
}

// ---------- stage 4: ||y||^2 = N*(sum m^2 + 0.5*m0^2); rinv = 1/(||y||*N) ----------
__global__ __launch_bounds__(256) void k_ostats(const _Float16* __restrict__ mixed_h,
                                                float* __restrict__ rinvo) {
    int row = blockIdx.x * 256 + threadIdx.x;
    const v8h* mr = (const v8h*)(mixed_h + (size_t)row * NMODES);
    float ss = 0.f;
    float m0 = 0.f;
#pragma unroll
    for (int c = 0; c < NMODES / 8; ++c) {
        v8h v = mr[c];
#pragma unroll
        for (int e = 0; e < 8; ++e) {
            float f = (float)v[e];
            ss += f * f;
            if (c == 0 && e == 0) m0 = f;
        }
    }
    float nrm = sqrtf(8192.0f * (ss + 0.5f * m0 * m0));
    rinvo[row] = (nrm > 0.0f) ? (1.0f / (nrm * 8192.0f)) : 0.0f;
}

// ---------- stage 5: out[row][n] = rinvo[row] * sum_k mixed[row][k] * M[n][k] ----------
// WG tile: 64 rows x 128 cols, K = 128 (4 unrolled steps).
__global__ __launch_bounds__(256) void k_stageC(const _Float16* __restrict__ mixed_h,
                                                const _Float16* __restrict__ BC,
                                                const float* __restrict__ rinvo,
                                                float* __restrict__ out) {
    int wave = threadIdx.x >> 5;
    int lane = threadIdx.x & 31;
    int half = lane >> 4;
    int l    = lane & 15;
    int mt = wave >> 1;
    int nh = wave & 1;
    int rowbase = blockIdx.x * 64;
    int nbase   = blockIdx.y * 128;
    int arow = rowbase + mt * 16 + l;
    const _Float16* ap = mixed_h + (size_t)arow * NMODES;

    v8f acc[4] = {};
#pragma unroll
    for (int kk = 0; kk < 4; ++kk) {
        int c0 = kk * 32 + half * 8;
        v16h a = cat8(*(const v8h*)(ap + c0), *(const v8h*)(ap + c0 + 16));
#pragma unroll
        for (int j = 0; j < 4; ++j) {
            int n = nbase + (nh * 4 + j) * 16 + l;
            v16h b = cat8(*(const v8h*)(BC + ((size_t)kk * NLEN + n) * 32 + half * 8), *(const v8h*)(BC + ((size_t)kk * NLEN + n) * 32 + half * 8 + 16));
            acc[j] = __builtin_amdgcn_wmma_f32_16x16x32_f16(
                false, a, false, b, (short)0, acc[j], false, false);
        }
    }
#pragma unroll
    for (int j = 0; j < 4; ++j) {
        int n = nbase + (nh * 4 + j) * 16 + l;
#pragma unroll
        for (int e = 0; e < 8; ++e) {
            int row = rowbase + mt * 16 + e + half * 8;
            out[(size_t)row * NLEN + n] = acc[j][e] * rinvo[row];
        }
    }
}

extern "C" void kernel_launch(void* const* d_in, const int* in_sizes, int n_in,
                              void* d_out, int out_size, void* d_ws, size_t ws_size,
                              hipStream_t stream) {
    (void)in_sizes; (void)n_in; (void)out_size; (void)ws_size;
    const float* x  = (const float*)d_in[0];   // [64,128,8192] f32
    const float* w1 = (const float*)d_in[1];   // [128,128,128] f32
    float* out = (float*)d_out;                // [64,128,8192] f32

    char* ws = (char*)d_ws;
    float*    rinvx   = (float*)(ws + NORMX_OFF);
    float*    rinvo   = (float*)(ws + RINVO_OFF);
    _Float16* BA      = (_Float16*)(ws + BA_OFF);
    _Float16* BC      = (_Float16*)(ws + BC_OFF);
    _Float16* X1h     = (_Float16*)(ws + X1H_OFF);
    _Float16* Wp      = (_Float16*)(ws + WP_OFF);
    _Float16* Wm      = (_Float16*)(ws + WM_OFF);
    _Float16* mixed_h = (_Float16*)(ws + MIXH_OFF);

    k_build_BA<<<4096, 256, 0, stream>>>(BA);
    k_build_BC<<<4096, 256, 0, stream>>>(BC);
    k_build_W <<<8192, 256, 0, stream>>>(w1, Wp, Wm);
    k_norms   <<<NROWS, 256, 0, stream>>>(x, rinvx);
    k_stageA  <<<NROWS / 16, 256, 0, stream>>>(x, BA, rinvx, X1h);
    k_stageB  <<<NMODES, 256, 0, stream>>>(X1h, Wp, Wm, mixed_h);
    k_ostats  <<<NROWS / 256, 256, 0, stream>>>(mixed_h, rinvo);
    dim3 gC(NROWS / 64, NLEN / 128);
    k_stageC  <<<gC, 256, 0, stream>>>(mixed_h, BC, rinvo, out);
}
